// MultiHeadSelfAttention_72353019068943
// MI455X (gfx1250) — matrix operand order repaired and checked by reading, not compiled, not run on hardware
//
#include <hip/hip_runtime.h>
#include <hip/hip_bf16.h>
#include <stdint.h>

// ---------------- constants (match reference) ----------------
static constexpr int Bc = 4, Hc = 16, Sc = 2048, Dc = 64, Ec = 1024;

// ---------------- WMMA plumbing ----------------
typedef __attribute__((ext_vector_type(16))) __bf16 v16bf;
typedef __attribute__((ext_vector_type(8)))  float  v8f;

union FragB16 { v16bf v; uint32_t u[8]; };

static __device__ __forceinline__ v8f wmma_bf16(const uint32_t (&a)[8],
                                                const uint32_t (&b)[8], v8f c) {
  FragB16 A, Bf;
#pragma unroll
  for (int i = 0; i < 8; ++i) { A.u[i] = a[i]; Bf.u[i] = b[i]; }
  return __builtin_amdgcn_wmma_f32_16x16x32_bf16(false, A.v, false, Bf.v,
                                                 (short)0, c, false, false);
}

static __device__ __forceinline__ uint16_t f2bf(float f) {
  uint32_t u = __builtin_bit_cast(uint32_t, f);
  u += 0x7FFFu + ((u >> 16) & 1u);   // round-to-nearest-even
  return (uint16_t)(u >> 16);
}

// ---------------- 16-lane (DPP row) butterfly reductions, pure VALU ----------------
static __device__ __forceinline__ float row16_max(float x) {
  x = fmaxf(x, __builtin_bit_cast(float, __builtin_amdgcn_update_dpp(
                   0, __builtin_bit_cast(int, x), 0xB1, 0xF, 0xF, true)));  // quad xor1
  x = fmaxf(x, __builtin_bit_cast(float, __builtin_amdgcn_update_dpp(
                   0, __builtin_bit_cast(int, x), 0x4E, 0xF, 0xF, true)));  // quad xor2
  x = fmaxf(x, __builtin_bit_cast(float, __builtin_amdgcn_update_dpp(
                   0, __builtin_bit_cast(int, x), 0x141, 0xF, 0xF, true))); // half-mirror
  x = fmaxf(x, __builtin_bit_cast(float, __builtin_amdgcn_update_dpp(
                   0, __builtin_bit_cast(int, x), 0x140, 0xF, 0xF, true))); // row mirror
  return x;
}
static __device__ __forceinline__ float row16_sum(float x) {
  x += __builtin_bit_cast(float, __builtin_amdgcn_update_dpp(
           0, __builtin_bit_cast(int, x), 0xB1, 0xF, 0xF, true));
  x += __builtin_bit_cast(float, __builtin_amdgcn_update_dpp(
           0, __builtin_bit_cast(int, x), 0x4E, 0xF, 0xF, true));
  x += __builtin_bit_cast(float, __builtin_amdgcn_update_dpp(
           0, __builtin_bit_cast(int, x), 0x141, 0xF, 0xF, true));
  x += __builtin_bit_cast(float, __builtin_amdgcn_update_dpp(
           0, __builtin_bit_cast(int, x), 0x140, 0xF, 0xF, true));
  return x;
}

// ---------------- TDM: 2-D tensor tile -> LDS (clang-23 6-arg builtin) ----------------
typedef uint32_t u32x4_t __attribute__((ext_vector_type(4)));
typedef int      i32x8_t __attribute__((ext_vector_type(8)));
typedef int      i32x4_t __attribute__((ext_vector_type(4)));

// Loads a tile_d1 x tile_d0 (rows x elems, bf16) tile starting at gptr into LDS at
// lds_off, with pad_amount(code) dwords of LDS padding every 2^(pad_iv_code+1) dwords.
static __device__ __forceinline__ void tdm_load_2d(const uint16_t* gptr, uint32_t lds_off,
                                                   uint32_t tile_d0, uint32_t tile_d1,
                                                   uint32_t tensor_d0, uint32_t tensor_d1,
                                                   uint32_t stride0,
                                                   uint32_t pad_iv_code, uint32_t pad_amt_code) {
  const uint64_t ga = (uint64_t)(uintptr_t)gptr;
  u32x4_t g0;
  g0[0] = 1u;                                                   // count=1, user D#
  g0[1] = lds_off;                                              // LDS byte address
  g0[2] = (uint32_t)ga;                                         // global_addr[31:0]
  g0[3] = ((uint32_t)(ga >> 32) & 0x01FFFFFFu) | 0x80000000u;   // addr[56:32] | type=2
  i32x8_t g1;
  g1[0] = (int)((1u << 16)                      // data_size = 2B
                | (1u << 20)                    // pad_enable
                | (pad_iv_code << 22) | (pad_amt_code << 25));
  g1[1] = (int)(tensor_d0 << 16);                               // tensor_dim0[15:0]
  g1[2] = (int)((tensor_d0 >> 16) | (tensor_d1 << 16));         // dim0 hi | dim1 lo
  g1[3] = (int)((tensor_d1 >> 16) | (tile_d0 << 16));           // dim1 hi | tile_dim0
  g1[4] = (int)(tile_d1 & 0xFFFFu);                             // tile_dim1 (tile_dim2=0)
  g1[5] = (int)stride0;                                         // dim0_stride[31:0]
  g1[6] = 0;                                                    // stride hi | dim1_stride lo
  g1[7] = 0;
  const i32x4_t z4 = {0, 0, 0, 0};
  const i32x8_t z8 = {0, 0, 0, 0, 0, 0, 0, 0};
  __builtin_amdgcn_tensor_load_to_lds(g0, g1, z4, z4, z8, 0);
}

// ---------------- fp32 -> bf16 convert (vectorized x4) ----------------
__global__ __launch_bounds__(256) void cvt_f32_bf16x4(const float* __restrict__ in,
                                                      uint16_t* __restrict__ out,
                                                      int n4) {
  int i = blockIdx.x * 256 + threadIdx.x;
  if (i < n4) {
    float4 f = ((const float4*)in)[i];
    uint2 p;
    p.x = (uint32_t)f2bf(f.x) | ((uint32_t)f2bf(f.y) << 16);
    p.y = (uint32_t)f2bf(f.z) | ((uint32_t)f2bf(f.w) << 16);
    ((uint2*)out)[i] = p;
  }
}

// ---------------- bf16 WMMA GEMM: [8192,1024] x [1024,1024] + bias ----------------
// Block tile 128x64, 8 waves as 4(M) x 2(N), each wave 2x2 16x16 WMMA tiles, BK=32.
// A tile staged by the Tensor Data Mover; B tile transposed manually.
// MODE 0: write bf16 into [b,h,s,d] head-split layout; MODE 1: fp32 row-major.
template <int MODE>
__global__ __launch_bounds__(256) void gemm_bf16_k(const uint16_t* __restrict__ A,
                                                   const uint16_t* __restrict__ Bw,
                                                   const float* __restrict__ bias,
                                                   void* __restrict__ OutP) {
  constexpr int Kdim = Ec, N = Ec, M = Bc * Sc;
  __shared__ uint16_t sA[128][40];   // 64B data + 16B TDM pad per row
  __shared__ uint16_t sBT[64][40];   // B transposed: [n][k], k-pairs contiguous
  const int tid = threadIdx.x, wid = tid >> 5, lane = tid & 31;
  const int half = lane >> 4, lm = lane & 15;
  const int wm = wid & 3, wn = wid >> 2;
  const int row0 = blockIdx.y * 128, col0 = blockIdx.x * 64;
  v8f acc[2][2] = {};
  const int bk_ = tid >> 3, bc = (tid & 7) * 8;          // B: 8 thr/row, 8 bf16 each
  const uint32_t sA_off = (uint32_t)(uintptr_t)&sA[0][0];
  for (int kb = 0; kb < Kdim; kb += 32) {
    __syncthreads();
    if (wid == 0) {  // TDM: 128-row x 32-elem A tile -> LDS (pad 4dw / 16dw)
      tdm_load_2d(A + (size_t)row0 * Kdim + kb, sA_off,
                  /*tile*/ 32, 128, /*tensor*/ Kdim, M, /*stride*/ Kdim, 3, 3);
    }
    {  // stage B tile 32x64, transposed into sBT
      const uint16_t* bg = Bw + (size_t)(kb + bk_) * N + col0 + bc;
      uint4 q = *(const uint4*)bg;
      sBT[bc + 0][bk_] = (uint16_t)q.x; sBT[bc + 1][bk_] = (uint16_t)(q.x >> 16);
      sBT[bc + 2][bk_] = (uint16_t)q.y; sBT[bc + 3][bk_] = (uint16_t)(q.y >> 16);
      sBT[bc + 4][bk_] = (uint16_t)q.z; sBT[bc + 5][bk_] = (uint16_t)(q.z >> 16);
      sBT[bc + 6][bk_] = (uint16_t)q.w; sBT[bc + 7][bk_] = (uint16_t)(q.w >> 16);
    }
    if (wid == 0) __builtin_amdgcn_s_wait_tensorcnt(0);
    __syncthreads();
    uint32_t af[2][8], bf[2][8];
#pragma unroll
    for (int mi = 0; mi < 2; ++mi) {
      const int r = wm * 32 + mi * 16 + lm;   // A layout: m = lane&15
#pragma unroll
      for (int v = 0; v < 8; ++v) {
        const int k0 = ((v & 4) ? 16 : 0) + half * 8 + (v & 3) * 2;
        af[mi][v] = *(const uint32_t*)&sA[r][k0];
      }
    }
#pragma unroll
    for (int ni = 0; ni < 2; ++ni) {
      const int ccol = wn * 32 + ni * 16 + lm;  // B layout: n = lane&15
#pragma unroll
      for (int v = 0; v < 8; ++v) {
        const int k0 = ((v & 4) ? 16 : 0) + half * 8 + (v & 3) * 2;
        bf[ni][v] = *(const uint32_t*)&sBT[ccol][k0];
      }
    }
#pragma unroll
    for (int mi = 0; mi < 2; ++mi)
#pragma unroll
      for (int ni = 0; ni < 2; ++ni)
        acc[mi][ni] = wmma_bf16(af[mi], bf[ni], acc[mi][ni]);
  }
  // epilogue: C layout m = vgpr + 8*half, n = lane&15
#pragma unroll
  for (int mi = 0; mi < 2; ++mi) {
#pragma unroll
    for (int ni = 0; ni < 2; ++ni) {
#pragma unroll
      for (int v = 0; v < 8; ++v) {
        const int r = row0 + wm * 32 + mi * 16 + half * 8 + v;
        const int n = col0 + wn * 32 + ni * 16 + lm;
        const float val = acc[mi][ni][v] + bias[n];
        if (MODE == 0) {
          uint16_t* Out = (uint16_t*)OutP;
          const int b = r >> 11, s = r & (Sc - 1);
          const int h = n >> 6, d = n & 63;
          Out[((((size_t)b * Hc + h) * Sc + s) << 6) + d] = f2bf(val);
        } else {
          ((float*)OutP)[(size_t)r * N + n] = val;
        }
      }
    }
  }
}

// ---------------- flash attention: one (b,h) per blockIdx.y, 128 q-rows/block ----------------
// K tiles double-buffered through the TDM (prefetch one 32-key block ahead);
// V staged transposed by the waves; softmax row reductions via DPP.
__global__ __launch_bounds__(256) void attn_k(const uint16_t* __restrict__ Qb,
                                              const uint16_t* __restrict__ Kb,
                                              const uint16_t* __restrict__ Vb,
                                              uint16_t* __restrict__ ctx) {
  __shared__ uint16_t sK[2][32][72];    // [buf][key][d], 128B data + 16B TDM pad per row
  __shared__ uint16_t sVT[64][40];      // [d][key] transposed (PV B-operand pairs along key)
  __shared__ uint16_t sP[8][16][32];    // per-wave C->A layout transpose patch
  const int tid = threadIdx.x, wid = tid >> 5, lane = tid & 31;
  const int half = lane >> 4, lm = lane & 15;
  const int bh = blockIdx.y, qblk = blockIdx.x;
  const float scale = 0.125f;  // 1/sqrt(64)
  const uint16_t* Qp = Qb + (size_t)bh * Sc * Dc;
  const uint16_t* Kp = Kb + (size_t)bh * Sc * Dc;
  const uint16_t* Vp = Vb + (size_t)bh * Sc * Dc;

  // Q fragments for this wave's 16 rows, held in VGPRs all kernel (A-operand layout)
  uint32_t aq[2][8];
  const int qrow = qblk * 128 + wid * 16 + lm;
#pragma unroll
  for (int dstep = 0; dstep < 2; ++dstep)
#pragma unroll
    for (int v = 0; v < 8; ++v) {
      const int k0 = dstep * 32 + ((v & 4) ? 16 : 0) + half * 8 + (v & 3) * 2;
      aq[dstep][v] = *(const uint32_t*)&Qp[(size_t)qrow * Dc + k0];
    }

  float m_run[8], l_run[8];
#pragma unroll
  for (int v = 0; v < 8; ++v) { m_run[v] = -3.0e38f; l_run[v] = 0.f; }
  v8f O[4] = {};

  const int tkey = tid >> 3, tc = (tid & 7) * 8;

  // prefetch first K tile (32 keys x 64 d, pad 4dw / 32dw)
  if (wid == 0) {
    tdm_load_2d(Kp, (uint32_t)(uintptr_t)&sK[0][0][0],
                /*tile*/ Dc, 32, /*tensor*/ Dc, Sc, /*stride*/ Dc, 4, 3);
  }

  for (int kb0 = 0; kb0 < Sc; kb0 += 32) {
    const int cur = (kb0 >> 5) & 1;
    __syncthreads();  // all reads of both LDS buffers from previous iteration done
    if (wid == 0 && kb0 + 32 < Sc) {  // TDM prefetch of next K tile into other buffer
      tdm_load_2d(Kp + (size_t)(kb0 + 32) * Dc, (uint32_t)(uintptr_t)&sK[cur ^ 1][0][0],
                  Dc, 32, Dc, Sc, Dc, 4, 3);
    }
    {  // stage V tile transposed (all waves)
      uint4 vq = *(const uint4*)&Vp[(size_t)(kb0 + tkey) * Dc + tc];
      sVT[tc + 0][tkey] = (uint16_t)vq.x; sVT[tc + 1][tkey] = (uint16_t)(vq.x >> 16);
      sVT[tc + 2][tkey] = (uint16_t)vq.y; sVT[tc + 3][tkey] = (uint16_t)(vq.y >> 16);
      sVT[tc + 4][tkey] = (uint16_t)vq.z; sVT[tc + 5][tkey] = (uint16_t)(vq.z >> 16);
      sVT[tc + 6][tkey] = (uint16_t)vq.w; sVT[tc + 7][tkey] = (uint16_t)(vq.w >> 16);
    }
    if (wid == 0) {  // wait: current K tile resident (allow 1 outstanding prefetch)
      if (kb0 + 32 < Sc) __builtin_amdgcn_s_wait_tensorcnt(1);
      else               __builtin_amdgcn_s_wait_tensorcnt(0);
    }
    __syncthreads();

    // scores: two 16x16 tiles (keys j*16..j*16+15), contraction d=64 chained over 2 WMMA
    v8f sc2[2];
#pragma unroll
    for (int j = 0; j < 2; ++j) {
      v8f c = {};
#pragma unroll
      for (int dstep = 0; dstep < 2; ++dstep) {
        uint32_t bk[8];
#pragma unroll
        for (int v = 0; v < 8; ++v) {
          const int dd = dstep * 32 + ((v & 4) ? 16 : 0) + half * 8 + (v & 3) * 2;
          bk[v] = *(const uint32_t*)&sK[cur][j * 16 + lm][dd];
        }
        c = wmma_bf16(aq[dstep], bk, c);
      }
      sc2[j] = c;
    }

    // online softmax: row m = v + 8*half lives across the 16 lanes of a half (DPP row)
    float alpha[8];
#pragma unroll
    for (int v = 0; v < 8; ++v) {
      float x0 = sc2[0][v] * scale, x1 = sc2[1][v] * scale;
      const float mx = row16_max(fmaxf(x0, x1));
      const float mn = fmaxf(m_run[v], mx);
      const float a  = __expf(m_run[v] - mn);
      const float p0 = __expf(x0 - mn), p1 = __expf(x1 - mn);
      const float rs = row16_sum(p0 + p1);
      l_run[v] = l_run[v] * a + rs;
      m_run[v] = mn;
      alpha[v] = a;
      sP[wid][half * 8 + v][lm]      = f2bf(p0);   // C-layout -> LDS
      sP[wid][half * 8 + v][16 + lm] = f2bf(p1);
    }
#pragma unroll
    for (int t = 0; t < 4; ++t)
#pragma unroll
      for (int v = 0; v < 8; ++v) O[t][v] *= alpha[v];

    // reload P in A-operand layout (wave-private LDS region; DS ops in-order per wave)
    uint32_t pa[8];
#pragma unroll
    for (int v = 0; v < 8; ++v) {
      const int k0 = ((v & 4) ? 16 : 0) + half * 8 + (v & 3) * 2;
      pa[v] = *(const uint32_t*)&sP[wid][lm][k0];
    }
    // O += P(16x32) x V(32x64): 4 d-tiles
#pragma unroll
    for (int t = 0; t < 4; ++t) {
      uint32_t bv[8];
#pragma unroll
      for (int v = 0; v < 8; ++v) {
        const int k0 = ((v & 4) ? 16 : 0) + half * 8 + (v & 3) * 2;
        bv[v] = *(const uint32_t*)&sVT[t * 16 + lm][k0];
      }
      O[t] = wmma_bf16(pa, bv, O[t]);
    }
  }

  // epilogue: normalize and scatter into ctx [b, s, h*64+d] (bf16)
  const int b = bh >> 4, h = bh & 15;
#pragma unroll
  for (int v = 0; v < 8; ++v) {
    const float inv = 1.0f / l_run[v];
    const int srow = qblk * 128 + wid * 16 + half * 8 + v;
#pragma unroll
    for (int t = 0; t < 4; ++t) {
      ctx[((size_t)b * Sc + srow) * Ec + h * 64 + t * 16 + lm] = f2bf(O[t][v] * inv);
    }
  }
}

// ---------------- host-side orchestration ----------------
extern "C" void kernel_launch(void* const* d_in, const int* in_sizes, int n_in,
                              void* d_out, int out_size, void* d_ws, size_t ws_size,
                              hipStream_t stream) {
  (void)in_sizes; (void)n_in; (void)out_size; (void)ws_size;
  const float* x  = (const float*)d_in[0];
  const float* Wq = (const float*)d_in[1];
  const float* bq = (const float*)d_in[2];
  const float* Wk = (const float*)d_in[3];
  const float* bk = (const float*)d_in[4];
  const float* Wv = (const float*)d_in[5];
  const float* bv = (const float*)d_in[6];
  const float* Wo = (const float*)d_in[7];
  const float* bo = (const float*)d_in[8];

  const size_t NX = (size_t)Bc * Sc * Ec;  // 8,388,608
  const size_t NW = (size_t)Ec * Ec;       // 1,048,576
  char* ws = (char*)d_ws;
  uint16_t* xb  = (uint16_t*)ws;                       // 16 MB
  uint16_t* wqb = (uint16_t*)(ws + 2 * NX);            // 2 MB each
  uint16_t* wkb = wqb + NW;
  uint16_t* wvb = wkb + NW;
  uint16_t* wob = wvb + NW;
  uint16_t* Qb  = (uint16_t*)(ws + 2 * NX + 8 * NW);   // 16 MB each, [b,h,s,d]
  uint16_t* Kb  = Qb + NX;
  uint16_t* Vb  = Kb + NX;
  uint16_t* ctx = Vb + NX;                             // 16 MB, [b,s,e]

  // 1) fp32 -> bf16
  cvt_f32_bf16x4<<<(int)(NX / 1024), 256, 0, stream>>>(x,  xb,  (int)(NX / 4));
  cvt_f32_bf16x4<<<(int)(NW / 1024), 256, 0, stream>>>(Wq, wqb, (int)(NW / 4));
  cvt_f32_bf16x4<<<(int)(NW / 1024), 256, 0, stream>>>(Wk, wkb, (int)(NW / 4));
  cvt_f32_bf16x4<<<(int)(NW / 1024), 256, 0, stream>>>(Wv, wvb, (int)(NW / 4));
  cvt_f32_bf16x4<<<(int)(NW / 1024), 256, 0, stream>>>(Wo, wob, (int)(NW / 4));

  // 2) Q/K/V projections (head-split bf16 outputs)
  dim3 gg(Ec / 64, (Bc * Sc) / 128);  // (16, 64)
  gemm_bf16_k<0><<<gg, 256, 0, stream>>>(xb, wqb, bq, Qb);
  gemm_bf16_k<0><<<gg, 256, 0, stream>>>(xb, wkb, bk, Kb);
  gemm_bf16_k<0><<<gg, 256, 0, stream>>>(xb, wvb, bv, Vb);

  // 3) flash attention
  attn_k<<<dim3(Sc / 128, Bc * Hc), 256, 0, stream>>>(Qb, Kb, Vb, ctx);

  // 4) output projection (fp32 out + bias)
  gemm_bf16_k<1><<<gg, 256, 0, stream>>>(ctx, wob, bo, d_out);
}
